// NeuronMiniMaxM2DecoderLayer_88880053224132
// MI455X (gfx1250) — matrix operand order repaired and checked by reading, not compiled, not run on hardware
//
#include <hip/hip_runtime.h>
#include <hip/hip_bf16.h>

#define S_   1024
#define H_   2048
#define NH_  16
#define NKV_ 4
#define HD_  128
#define RD_  64
#define E_   16
#define I_   1024
#define MPAD (2 * S_ + 64)   // padded row count for expert-gathered matrices

typedef __bf16 v16bf __attribute__((ext_vector_type(16)));
typedef __bf16 v8bf  __attribute__((ext_vector_type(8)));
typedef float  v8f   __attribute__((ext_vector_type(8)));

__device__ __forceinline__ v8f wmma_bf16(v16bf a, v16bf b, v8f c) {
  return __builtin_amdgcn_wmma_f32_16x16x32_bf16(false, a, false, b, (short)0, c, false, false);
}
__device__ __forceinline__ void zero8(v8f& v) {
#pragma unroll
  for (int i = 0; i < 8; ++i) v[i] = 0.0f;
}
__device__ __forceinline__ v16bf cat8(v8bf lo, v8bf hi) {
  return __builtin_shufflevector(lo, hi, 0, 1, 2, 3, 4, 5, 6, 7, 8, 9, 10, 11, 12, 13, 14, 15);
}
// A-fragment (16x32 bf16): two contiguous 8-elem k-runs at kin = half*8 and 16+half*8
__device__ __forceinline__ v16bf load_afrag(const __bf16* __restrict__ Asw, size_t Ms,
                                            int kb, int row, int half) {
  const __bf16* p = Asw + ((size_t)kb * Ms + row) * 32;
  v8bf lo = *(const v8bf*)(const void*)(p + half * 8);
  v8bf hi = *(const v8bf*)(const void*)(p + 16 + half * 8);
  return cat8(lo, hi);
}
// B-fragment (32x16 bf16): two contiguous 8-elem k-runs at kin = half*8 and 16+half*8
__device__ __forceinline__ v16bf load_bfrag(const __bf16* __restrict__ Bsw, size_t N,
                                            int kb, int n, int half) {
  const __bf16* p = Bsw + ((size_t)kb * N + n) * 32;
  v8bf lo = *(const v8bf*)(const void*)(p + half * 8);
  v8bf hi = *(const v8bf*)(const void*)(p + 16 + half * 8);
  return cat8(lo, hi);
}

// ---------------------------------------------------------------------------
// Weight pre-swizzle: f32 [K][N] row-major -> bf16 [K/32][N][32]
// ---------------------------------------------------------------------------
__global__ __launch_bounds__(256) void k_swizzle_w(const float* __restrict__ B,
                                                   __bf16* __restrict__ out, int N) {
  const long u = (long)blockIdx.x * 256 + threadIdx.x;
  const int n = (int)(u % N);
  const long k0 = (u / N) * 8;
  __align__(16) __bf16 tmp[8];
#pragma unroll
  for (int j = 0; j < 8; ++j) tmp[j] = (__bf16)B[(k0 + j) * N + n];
  __bf16* dst = out + ((k0 >> 5) * N + n) * 32 + (int)(k0 & 31);
  *(float4*)(void*)dst = *(const float4*)(const void*)tmp;
}

// ---------------------------------------------------------------------------
// RMSNorm (f32 in) -> bf16 out in swizzled A-layout [H/32][S][32]
// ---------------------------------------------------------------------------
__global__ __launch_bounds__(256) void k_rmsnorm_cast(const float* __restrict__ x,
                                                      const float* __restrict__ w,
                                                      __bf16* __restrict__ y) {
  const int t = blockIdx.x;
  const float* xr = x + (size_t)t * H_;
  __shared__ float red[256];
  float s = 0.f;
  for (int i = threadIdx.x; i < H_; i += 256) { float v = xr[i]; s += v * v; }
  red[threadIdx.x] = s; __syncthreads();
  for (int o = 128; o > 0; o >>= 1) { if (threadIdx.x < o) red[threadIdx.x] += red[threadIdx.x + o]; __syncthreads(); }
  const float rms = rsqrtf(red[0] / (float)H_ + 1e-6f);
  for (int i = threadIdx.x; i < H_; i += 256)
    y[((size_t)(i >> 5) * S_ + t) * 32 + (i & 31)] = (__bf16)(w[i] * xr[i] * rms);
}

// ---------------------------------------------------------------------------
// Register-resident GEMM: C[M,N](f32,row-major) = Asw * Bsw (+resid, compile-time)
// block 128 thr = 4 waves, tile 64x64, all fragments direct from global.
// ---------------------------------------------------------------------------
template <bool RESID>
__global__ __launch_bounds__(128) void k_gemm(const __bf16* __restrict__ Asw, int Ms,
                                              const __bf16* __restrict__ Bsw,
                                              float* __restrict__ C, int N,
                                              const float* __restrict__ resid, int K) {
  const int tid = threadIdx.x, lane = tid & 31, wave = tid >> 5;
  const int half = lane >> 4, l16 = lane & 15;
  const int mbase = blockIdx.y * 64, nbase = blockIdx.x * 64;
  const int row = mbase + wave * 16 + l16;
  v8f acc[4];
#pragma unroll
  for (int nt = 0; nt < 4; ++nt) zero8(acc[nt]);

  const int nkb = K >> 5;
#pragma unroll 2
  for (int kb = 0; kb < nkb; ++kb) {
    v16bf af = load_afrag(Asw, (size_t)Ms, kb, row, half);
#pragma unroll
    for (int nt = 0; nt < 4; ++nt) {
      v16bf bf = load_bfrag(Bsw, (size_t)N, kb, nbase + nt * 16 + l16, half);
      acc[nt] = wmma_bf16(af, bf, acc[nt]);
    }
  }
#pragma unroll
  for (int nt = 0; nt < 4; ++nt)
#pragma unroll
    for (int r = 0; r < 8; ++r) {
      int m = mbase + wave * 16 + half * 8 + r;
      int n = nbase + nt * 16 + l16;
      float v = acc[nt][r];
      if (RESID) v += resid[(size_t)m * N + n];
      C[(size_t)m * N + n] = v;
    }
}

// ---------------------------------------------------------------------------
// QK RMSNorm + partial RoPE; outputs in fragment-native layouts:
//   qsw[head][HD/32][S][32]  (A-layout per head)
//   ksw[kvh][HD/32][S][32]   (B-layout over keys: n=key, k=hd)
//   vsw[kvh][S/32][HD][32]   (B-layout over hd:   n=hd,  k=key)
// ---------------------------------------------------------------------------
__global__ __launch_bounds__(256) void k_qknorm_rope(const float* __restrict__ qf,
                                                     const float* __restrict__ kf,
                                                     const float* __restrict__ vf,
                                                     const float* __restrict__ qn_w,
                                                     const float* __restrict__ kn_w,
                                                     const long long* __restrict__ pos,
                                                     __bf16* __restrict__ qsw,
                                                     __bf16* __restrict__ ksw,
                                                     __bf16* __restrict__ vsw) {
  const int t = blockIdx.x;
  __shared__ float red[256];
  __shared__ float rmsq_s, rmsk_s;
  const float* q = qf + (size_t)t * (NH_ * HD_);
  const float* k = kf + (size_t)t * (NKV_ * HD_);
  const float* v = vf + (size_t)t * (NKV_ * HD_);

  float s = 0.f;
  for (int i = threadIdx.x; i < NH_ * HD_; i += 256) { float x = q[i]; s += x * x; }
  red[threadIdx.x] = s; __syncthreads();
  for (int o = 128; o > 0; o >>= 1) { if (threadIdx.x < o) red[threadIdx.x] += red[threadIdx.x + o]; __syncthreads(); }
  if (threadIdx.x == 0) rmsq_s = rsqrtf(red[0] / (float)(NH_ * HD_) + 1e-6f);
  __syncthreads();
  s = 0.f;
  for (int i = threadIdx.x; i < NKV_ * HD_; i += 256) { float x = k[i]; s += x * x; }
  red[threadIdx.x] = s; __syncthreads();
  for (int o = 128; o > 0; o >>= 1) { if (threadIdx.x < o) red[threadIdx.x] += red[threadIdx.x + o]; __syncthreads(); }
  if (threadIdx.x == 0) rmsk_s = rsqrtf(red[0] / (float)(NKV_ * HD_) + 1e-6f);
  __syncthreads();
  const float p = (float)pos[t];

  for (int i = threadIdx.x; i < NH_ * HD_; i += 256) {
    int h = i >> 7, d = i & 127, base = h << 7;
    float val = qn_w[i] * q[i] * rmsq_s;
    if (d < RD_) {
      int j = d & 31;
      float inv = powf(10000.f, -(float)(2 * j) / (float)RD_);
      float c = cosf(p * inv), sn = sinf(p * inv);
      int d2 = (d < 32) ? d + 32 : d - 32;
      float vp = qn_w[base + d2] * q[base + d2] * rmsq_s;
      val = (d < 32) ? val * c - vp * sn : val * c + vp * sn;
    }
    qsw[(((size_t)h * 4 + (d >> 5)) * S_ + t) * 32 + (d & 31)] = (__bf16)val;
  }
  for (int i = threadIdx.x; i < NKV_ * HD_; i += 256) {
    int h = i >> 7, d = i & 127, base = h << 7;
    float val = kn_w[i] * k[i] * rmsk_s;
    if (d < RD_) {
      int j = d & 31;
      float inv = powf(10000.f, -(float)(2 * j) / (float)RD_);
      float c = cosf(p * inv), sn = sinf(p * inv);
      int d2 = (d < 32) ? d + 32 : d - 32;
      float vp = kn_w[base + d2] * k[base + d2] * rmsk_s;
      val = (d < 32) ? val * c - vp * sn : val * c + vp * sn;
    }
    ksw[(((size_t)h * 4 + (d >> 5)) * S_ + t) * 32 + (d & 31)] = (__bf16)val;
    vsw[(((size_t)h * (S_ / 32) + (t >> 5)) * HD_ + d) * 32 + (t & 31)] = (__bf16)v[i];
  }
}

// ---------------------------------------------------------------------------
// Flash attention (GQA, causal). 128 thr = 4 waves, 64 q-rows, key blocks of 32.
// Fragments direct from global; only the P transpose uses (wave-private) LDS.
// ---------------------------------------------------------------------------
__global__ __launch_bounds__(128) void k_attn(const __bf16* __restrict__ qsw,
                                              const __bf16* __restrict__ ksw,
                                              const __bf16* __restrict__ vsw,
                                              __bf16* __restrict__ ctx) {
  const int head = blockIdx.y, kvh = head >> 2;
  const int qbase = blockIdx.x * 64;
  const int tid = threadIdx.x, lane = tid & 31, wave = tid >> 5;
  const int half = lane >> 4, l16 = lane & 15;

  __shared__ __bf16 lP[4][16][32];

  const int qrow = qbase + wave * 16 + l16;
  v16bf aq[4];
#pragma unroll
  for (int kt = 0; kt < 4; ++kt)
    aq[kt] = load_afrag(qsw + (size_t)head * 4 * S_ * 32, (size_t)S_, kt, qrow, half);

  v8f co[8];
#pragma unroll
  for (int j = 0; j < 8; ++j) zero8(co[j]);
  float mrow[8], lrow[8];
#pragma unroll
  for (int r = 0; r < 8; ++r) { mrow[r] = -1e30f; lrow[r] = 0.f; }
  const float scale = 0.08838834764831845f;  // 1/sqrt(128)

  const __bf16* Kh = ksw + (size_t)kvh * 4 * S_ * 32;
  const __bf16* Vh = vsw + (size_t)kvh * (S_ / 32) * HD_ * 32;

  const int nkb = (qbase >> 5) + 2;
  for (int kb = 0; kb < nkb; ++kb) {
    const int kstart = kb * 32;
    v8f sacc[2]; zero8(sacc[0]); zero8(sacc[1]);
#pragma unroll
    for (int nt = 0; nt < 2; ++nt) {
      const int key = kstart + nt * 16 + l16;
#pragma unroll
      for (int kt = 0; kt < 4; ++kt) {
        v16bf bk = load_bfrag(Kh, (size_t)S_, kt, key, half);
        sacc[nt] = wmma_bf16(aq[kt], bk, sacc[nt]);
      }
    }

    float pv0[8], pv1[8];
#pragma unroll
    for (int r = 0; r < 8; ++r) {
      const int qr = qbase + wave * 16 + half * 8 + r;
      const bool m0 = (kstart + l16) > qr, m1 = (kstart + 16 + l16) > qr;
      float s0 = m0 ? -1e30f : sacc[0][r] * scale;
      float s1 = m1 ? -1e30f : sacc[1][r] * scale;
      float mx = fmaxf(s0, s1);
      for (int o = 8; o > 0; o >>= 1) mx = fmaxf(mx, __shfl_xor(mx, o, 16));
      const float mnew = fmaxf(mrow[r], mx);
      const float alpha = expf(mrow[r] - mnew);
      const float p0 = m0 ? 0.f : expf(s0 - mnew);
      const float p1 = m1 ? 0.f : expf(s1 - mnew);
      float rs = p0 + p1;
      for (int o = 8; o > 0; o >>= 1) rs += __shfl_xor(rs, o, 16);
      lrow[r] = lrow[r] * alpha + rs;
      mrow[r] = mnew;
#pragma unroll
      for (int j = 0; j < 8; ++j) co[j][r] *= alpha;
      pv0[r] = p0; pv1[r] = p1;
    }
#pragma unroll
    for (int r = 0; r < 8; ++r) {
      lP[wave][half * 8 + r][l16]      = (__bf16)pv0[r];
      lP[wave][half * 8 + r][16 + l16] = (__bf16)pv1[r];
    }
    // wave-private LDS region; DS ops are in-order within a wave
    v8bf plo = *(const v8bf*)(const void*)&lP[wave][l16][half * 8];
    v8bf phi = *(const v8bf*)(const void*)&lP[wave][l16][16 + half * 8];
    v16bf ap = cat8(plo, phi);
#pragma unroll
    for (int j = 0; j < 8; ++j) {
      v16bf bv = load_bfrag(Vh + (size_t)kb * HD_ * 32, (size_t)HD_, 0, j * 16 + l16, half);
      co[j] = wmma_bf16(ap, bv, co[j]);
    }
  }
#pragma unroll
  for (int r = 0; r < 8; ++r) {
    const float inv = 1.f / lrow[r];
    const int qr = qbase + wave * 16 + half * 8 + r;
#pragma unroll
    for (int j = 0; j < 8; ++j) {
      const int c = head * HD_ + j * 16 + l16;
      ctx[((size_t)(c >> 5) * S_ + qr) * 32 + (c & 31)] = (__bf16)(co[j][r] * inv);
    }
  }
}

// ---------------------------------------------------------------------------
// Router: wave e computes expert-e logit on swizzled xt; thread 0 does top-2
// ---------------------------------------------------------------------------
__global__ __launch_bounds__(512) void k_router(const __bf16* __restrict__ xt,
                                                const float* __restrict__ rw,
                                                const float* __restrict__ rb,
                                                int* __restrict__ topk_idx,
                                                float* __restrict__ topk_w) {
  const int t = blockIdx.x;
  const int wave = threadIdx.x >> 5, lane = threadIdx.x & 31;
  __shared__ float aff[E_];
  const float* w = rw + (size_t)wave * H_;
  float s = 0.f;
  for (int j = 0; j < H_ / 32; ++j)
    s += (float)xt[((size_t)j * S_ + t) * 32 + lane] * w[j * 32 + lane];
  for (int o = 16; o > 0; o >>= 1) s += __shfl_xor(s, o, 32);
  if (lane == 0) aff[wave] = 1.f / (1.f + expf(-s));
  __syncthreads();
  if (threadIdx.x == 0) {
    float b1 = -1e30f; int i1 = 0;
    for (int e = 0; e < E_; ++e) { float sc = aff[e] + rb[e]; if (sc > b1) { b1 = sc; i1 = e; } }
    float b2 = -1e30f; int i2 = 0;
    for (int e = 0; e < E_; ++e) { if (e == i1) continue; float sc = aff[e] + rb[e]; if (sc > b2) { b2 = sc; i2 = e; } }
    float w0 = aff[i1], w1 = aff[i2], inv = 1.f / (w0 + w1);
    topk_idx[t * 2] = i1; topk_idx[t * 2 + 1] = i2;
    topk_w[t * 2] = w0 * inv; topk_w[t * 2 + 1] = w1 * inv;
  }
}

// ---------------------------------------------------------------------------
// Expert assignment compaction (single block; LDS atomics; values are
// order-independent: each token reads back via its own slot)
// ---------------------------------------------------------------------------
__global__ __launch_bounds__(1024) void k_assign(const int* __restrict__ topk_idx,
                                                 int* __restrict__ cnt_g, int* __restrict__ off_g,
                                                 int* __restrict__ assign_token,
                                                 int* __restrict__ token_slot) {
  __shared__ int cnt[E_], off[E_], cur[E_];
  const int t = threadIdx.x;
  if (t < E_) cnt[t] = 0;
  __syncthreads();
  const int e0 = topk_idx[t * 2], e1 = topk_idx[t * 2 + 1];
  atomicAdd(&cnt[e0], 1); atomicAdd(&cnt[e1], 1);
  __syncthreads();
  if (t == 0) { int run = 0; for (int e = 0; e < E_; ++e) { off[e] = run; run += cnt[e]; } }
  __syncthreads();
  if (t < E_) { cnt_g[t] = cnt[t]; off_g[t] = off[t]; cur[t] = off[t]; }
  __syncthreads();
  int a0 = atomicAdd(&cur[e0], 1); assign_token[a0] = t; token_slot[t * 2] = a0;
  int a1 = atomicAdd(&cur[e1], 1); assign_token[a1] = t; token_slot[t * 2 + 1] = a1;
}

// Gather swizzled token rows into expert-grouped swizzled rows (Ms: S_ -> MPAD)
__global__ __launch_bounds__(256) void k_gather(const __bf16* __restrict__ xt,
                                                const int* __restrict__ assign_token,
                                                __bf16* __restrict__ xg) {
  const int a = blockIdx.x, t = assign_token[a];
  const int kb = threadIdx.x >> 2, part = (threadIdx.x & 3) * 8;  // 64 kb * 4 chunks
  *(float4*)(void*)(xg + ((size_t)kb * MPAD + a) * 32 + part) =
      *(const float4*)(const void*)(xt + ((size_t)kb * S_ + t) * 32 + part);
}

// ---------------------------------------------------------------------------
// MoE gate/up fused GEMMs + SiLU*mul -> act (bf16, swizzled A-layout, Ms=MPAD)
// ---------------------------------------------------------------------------
__global__ __launch_bounds__(128) void k_moe_gateup(const __bf16* __restrict__ xg,
                                                    const __bf16* __restrict__ wg,
                                                    const __bf16* __restrict__ wu,
                                                    const int* __restrict__ cnt_g,
                                                    const int* __restrict__ off_g,
                                                    __bf16* __restrict__ act) {
  const int e = blockIdx.z;
  const int cnt = cnt_g[e], off = off_g[e];
  const int mb = blockIdx.y * 64;
  if (mb >= cnt) return;
  const int nb = blockIdx.x * 64;
  const __bf16* Bg = wg + (size_t)e * H_ * I_;
  const __bf16* Bu = wu + (size_t)e * H_ * I_;
  const int tid = threadIdx.x, lane = tid & 31, wave = tid >> 5;
  const int half = lane >> 4, l16 = lane & 15;
  const int row = off + mb + wave * 16 + l16;   // rows >= cnt read garbage, discarded
  v8f ag[4], au[4];
#pragma unroll
  for (int nt = 0; nt < 4; ++nt) { zero8(ag[nt]); zero8(au[nt]); }

#pragma unroll 2
  for (int kb = 0; kb < H_ / 32; ++kb) {
    v16bf af = load_afrag(xg, (size_t)MPAD, kb, row, half);
#pragma unroll
    for (int nt = 0; nt < 4; ++nt) {
      v16bf b = load_bfrag(Bg, (size_t)I_, kb, nb + nt * 16 + l16, half);
      ag[nt] = wmma_bf16(af, b, ag[nt]);
      b = load_bfrag(Bu, (size_t)I_, kb, nb + nt * 16 + l16, half);
      au[nt] = wmma_bf16(af, b, au[nt]);
    }
  }
#pragma unroll
  for (int nt = 0; nt < 4; ++nt)
#pragma unroll
    for (int r = 0; r < 8; ++r) {
      int ml = mb + wave * 16 + half * 8 + r;
      if (ml < cnt) {
        float g = ag[nt][r], u = au[nt][r];
        float a = g / (1.f + expf(-g)) * u;               // silu(g)*u
        int c = nb + nt * 16 + l16;
        act[((size_t)(c >> 5) * MPAD + off + ml) * 32 + (c & 31)] = (__bf16)a;
      }
    }
}

// MoE down projection: eo[row][H] (f32) = act @ w_down[e]
__global__ __launch_bounds__(128) void k_moe_down(const __bf16* __restrict__ act,
                                                  const __bf16* __restrict__ wd,
                                                  const int* __restrict__ cnt_g,
                                                  const int* __restrict__ off_g,
                                                  float* __restrict__ eo) {
  const int e = blockIdx.z;
  const int cnt = cnt_g[e], off = off_g[e];
  const int mb = blockIdx.y * 64;
  if (mb >= cnt) return;
  const int nb = blockIdx.x * 64;
  const __bf16* B = wd + (size_t)e * I_ * H_;
  const int tid = threadIdx.x, lane = tid & 31, wave = tid >> 5;
  const int half = lane >> 4, l16 = lane & 15;
  const int row = off + mb + wave * 16 + l16;
  v8f acc[4];
#pragma unroll
  for (int nt = 0; nt < 4; ++nt) zero8(acc[nt]);

#pragma unroll 2
  for (int kb = 0; kb < I_ / 32; ++kb) {
    v16bf af = load_afrag(act, (size_t)MPAD, kb, row, half);
#pragma unroll
    for (int nt = 0; nt < 4; ++nt) {
      v16bf b = load_bfrag(B, (size_t)H_, kb, nb + nt * 16 + l16, half);
      acc[nt] = wmma_bf16(af, b, acc[nt]);
    }
  }
#pragma unroll
  for (int nt = 0; nt < 4; ++nt)
#pragma unroll
    for (int r = 0; r < 8; ++r) {
      int ml = mb + wave * 16 + half * 8 + r;
      if (ml < cnt)
        eo[(size_t)(off + ml) * H_ + nb + nt * 16 + l16] = acc[nt][r];
    }
}

// Combine: out = resid2 + w0*eo[slot0] + w1*eo[slot1]
__global__ __launch_bounds__(256) void k_combine(const float* __restrict__ hs,
                                                 const float* __restrict__ eo,
                                                 const int* __restrict__ token_slot,
                                                 const float* __restrict__ topk_w,
                                                 float* __restrict__ out) {
  const int t = blockIdx.x;
  const int s0 = token_slot[t * 2], s1 = token_slot[t * 2 + 1];
  const float w0 = topk_w[t * 2], w1 = topk_w[t * 2 + 1];
  for (int h = threadIdx.x; h < H_; h += 256)
    out[(size_t)t * H_ + h] = hs[(size_t)t * H_ + h]
                            + w0 * eo[(size_t)s0 * H_ + h]
                            + w1 * eo[(size_t)s1 * H_ + h];
}

// ---------------------------------------------------------------------------
extern "C" void kernel_launch(void* const* d_in, const int* in_sizes, int n_in,
                              void* d_out, int out_size, void* d_ws, size_t ws_size,
                              hipStream_t stream) {
  const float*     hidden = (const float*)d_in[0];
  const long long* pos    = (const long long*)d_in[1];
  const float*     ln1_w  = (const float*)d_in[2];
  const float*     ln2_w  = (const float*)d_in[3];
  const float*     wq     = (const float*)d_in[4];
  const float*     wk     = (const float*)d_in[5];
  const float*     wv     = (const float*)d_in[6];
  const float*     wo     = (const float*)d_in[7];
  const float*     qn_w   = (const float*)d_in[8];
  const float*     kn_w   = (const float*)d_in[9];
  const float*     rw     = (const float*)d_in[10];
  const float*     rb     = (const float*)d_in[11];
  const float*     w_gate = (const float*)d_in[12];
  const float*     w_up   = (const float*)d_in[13];
  const float*     w_down = (const float*)d_in[14];
  float* out = (float*)d_out;

  char* wsb = (char*)d_ws;
  size_t o = 0;
  auto alloc = [&](size_t bytes) -> void* {
    void* p = wsb + o; o = (o + bytes + 255) & ~(size_t)255; return p;
  };
  __bf16* xb    = (__bf16*)alloc((size_t)S_ * H_ * 2);           // ln1 out (swizzled)
  float*  qf    = (float*) alloc((size_t)S_ * NH_ * HD_ * 4);
  float*  kf    = (float*) alloc((size_t)S_ * NKV_ * HD_ * 4);
  float*  vf    = (float*) alloc((size_t)S_ * NKV_ * HD_ * 4);
  __bf16* qsw   = (__bf16*)alloc((size_t)NH_ * S_ * HD_ * 2);
  __bf16* kswb  = (__bf16*)alloc((size_t)NKV_ * S_ * HD_ * 2);
  __bf16* vswb  = (__bf16*)alloc((size_t)NKV_ * S_ * HD_ * 2);
  __bf16* ctxsw = (__bf16*)alloc((size_t)S_ * NH_ * HD_ * 2);
  float*  hs    = (float*) alloc((size_t)S_ * H_ * 4);
  __bf16* xt    = (__bf16*)alloc((size_t)S_ * H_ * 2);           // ln2 out (swizzled)
  int*    tki   = (int*)   alloc((size_t)S_ * 2 * 4);
  float*  tkw   = (float*) alloc((size_t)S_ * 2 * 4);
  int*    cnt   = (int*)   alloc(E_ * 4);
  int*    off   = (int*)   alloc(E_ * 4);
  int*    atok  = (int*)   alloc((size_t)S_ * 2 * 4);
  int*    tslot = (int*)   alloc((size_t)S_ * 2 * 4);
  __bf16* xg    = (__bf16*)alloc((size_t)H_ * MPAD * 2);         // swizzled, Ms=MPAD
  __bf16* actb  = (__bf16*)alloc((size_t)I_ * MPAD * 2);         // swizzled, Ms=MPAD
  float*  eo    = (float*) alloc((size_t)2 * S_ * H_ * 4);
  __bf16* wq_s  = (__bf16*)alloc((size_t)H_ * NH_ * HD_ * 2);
  __bf16* wk_s  = (__bf16*)alloc((size_t)H_ * NKV_ * HD_ * 2);
  __bf16* wv_s  = (__bf16*)alloc((size_t)H_ * NKV_ * HD_ * 2);
  __bf16* wo_s  = (__bf16*)alloc((size_t)NH_ * HD_ * H_ * 2);
  __bf16* wg_s  = (__bf16*)alloc((size_t)E_ * H_ * I_ * 2);
  __bf16* wu_s  = (__bf16*)alloc((size_t)E_ * H_ * I_ * 2);
  __bf16* wd_s  = (__bf16*)alloc((size_t)E_ * I_ * H_ * 2);
  (void)ws_size; (void)n_in; (void)in_sizes; (void)out_size;

  // 0) weight pre-swizzle (f32 -> bf16, fragment-linear)
  k_swizzle_w<<<(H_ / 8) * (NH_ * HD_) / 256, 256, 0, stream>>>(wq, wq_s, NH_ * HD_);
  k_swizzle_w<<<(H_ / 8) * (NKV_ * HD_) / 256, 256, 0, stream>>>(wk, wk_s, NKV_ * HD_);
  k_swizzle_w<<<(H_ / 8) * (NKV_ * HD_) / 256, 256, 0, stream>>>(wv, wv_s, NKV_ * HD_);
  k_swizzle_w<<<(NH_ * HD_ / 8) * H_ / 256, 256, 0, stream>>>(wo, wo_s, H_);
  k_swizzle_w<<<(E_ * H_ / 8) * I_ / 256, 256, 0, stream>>>(w_gate, wg_s, I_);
  k_swizzle_w<<<(E_ * H_ / 8) * I_ / 256, 256, 0, stream>>>(w_up, wu_s, I_);
  k_swizzle_w<<<(E_ * I_ / 8) * H_ / 256, 256, 0, stream>>>(w_down, wd_s, H_);

  // 1) ln1 -> xb (bf16 swizzled)
  k_rmsnorm_cast<<<S_, 256, 0, stream>>>(hidden, ln1_w, xb);
  // 2) QKV projections (WMMA, no-resid epilogue)
  k_gemm<false><<<dim3((NH_ * HD_) / 64, S_ / 64), 128, 0, stream>>>(xb, S_, wq_s, qf, NH_ * HD_, nullptr, H_);
  k_gemm<false><<<dim3((NKV_ * HD_) / 64, S_ / 64), 128, 0, stream>>>(xb, S_, wk_s, kf, NKV_ * HD_, nullptr, H_);
  k_gemm<false><<<dim3((NKV_ * HD_) / 64, S_ / 64), 128, 0, stream>>>(xb, S_, wv_s, vf, NKV_ * HD_, nullptr, H_);
  // 3) QK norm + RoPE + fragment-native packing
  k_qknorm_rope<<<S_, 256, 0, stream>>>(qf, kf, vf, qn_w, kn_w, pos, qsw, kswb, vswb);
  // 4) flash attention (WMMA)
  k_attn<<<dim3(S_ / 64, NH_), 128, 0, stream>>>(qsw, kswb, vswb, ctxsw);
  // 5) output projection + residual (WMMA, fused resid add)
  k_gemm<true><<<dim3(H_ / 64, S_ / 64), 128, 0, stream>>>(ctxsw, S_, wo_s, hs, H_, hidden, NH_ * HD_);
  // 6) ln2 -> xt (bf16 swizzled)
  k_rmsnorm_cast<<<S_, 256, 0, stream>>>(hs, ln2_w, xt);
  // 7) router + top-2
  k_router<<<S_, 512, 0, stream>>>(xt, rw, rb, tki, tkw);
  // 8) assignment compaction + gather
  k_assign<<<1, 1024, 0, stream>>>(tki, cnt, off, atok, tslot);
  k_gather<<<2 * S_, 256, 0, stream>>>(xt, atok, xg);
  // 9) expert gate/up + SiLU*mul (WMMA)
  k_moe_gateup<<<dim3(I_ / 64, S_ / 64, E_), 128, 0, stream>>>(xg, wg_s, wu_s, cnt, off, actb);
  // 10) expert down projection (WMMA)
  k_moe_down<<<dim3(H_ / 64, S_ / 64, E_), 128, 0, stream>>>(actb, wd_s, cnt, off, eo);
  // 11) combine + residual
  k_combine<<<S_, 256, 0, stream>>>(hs, eo, tslot, tkw, out);
}
